// PModel_89352499626062
// MI455X (gfx1250) — matrix operand order repaired and checked by reading, not compiled, not run on hardware
//
#include <hip/hip_runtime.h>
#include <cstdint>
#include <cstddef>

// ---------------------------------------------------------------------------
// Problem constants (match reference)
// ---------------------------------------------------------------------------
#define NN_   4096
#define DD_   512
#define NHID_ 256
#define CC_   64
#define PP_   16
#define K1_   (PP_ * DD_)      // 8192
#define K2_   (PP_ * 2 * CC_)  // 2048
#define EPS_  1e-12f

typedef __attribute__((ext_vector_type(16))) _Float16 v16h;
typedef __attribute__((ext_vector_type(8)))  float    v8f;
typedef __attribute__((ext_vector_type(4)))  unsigned int v4u;
typedef __attribute__((ext_vector_type(8)))  int      v8i;
typedef __attribute__((ext_vector_type(4)))  int      v4i;

// GEMM tiling
#define BM  128
#define BN  128
#define BK  32
#define LDT 40   // padded LDS row stride (halves): 32 data + 8 pad

#if defined(__has_builtin)
#if __has_builtin(__builtin_amdgcn_tensor_load_to_lds)
#define HAVE_TDM 1
#endif
#endif
#ifndef HAVE_TDM
#define HAVE_TDM 0
#endif

union AFragU  { v16h v; unsigned int u[8]; };
union BFragU  { v16h v; uint4 q[2]; };
union Pack32B { uint4 q[2]; _Float16 h[16]; };

// ---------------------------------------------------------------------------
// Async copy of 32 bytes (16 halves) global -> LDS (ASYNCcnt tracked).
// INST_OFFSET is added to BOTH the LDS destination and the global source
// (ISA 08_async §4.4), so offsets 0/16 move the full 32B chunk.
// ---------------------------------------------------------------------------
__device__ __forceinline__ void async_cp32(const _Float16* g, void* lds) {
  const unsigned int l = (unsigned int)(uintptr_t)lds;
  asm volatile("global_load_async_to_lds_b128 %0, %1, off"
               ::"v"(l), "v"(g) : "memory");
  asm volatile("global_load_async_to_lds_b128 %0, %1, off offset:16"
               ::"v"(l), "v"(g) : "memory");
}
__device__ __forceinline__ void wait_async0() {
  asm volatile("s_wait_asynccnt 0x0" ::: "memory");
}
__device__ __forceinline__ void wait_tensor0() {
#if defined(__has_builtin)
#if __has_builtin(__builtin_amdgcn_s_wait_tensorcnt)
  __builtin_amdgcn_s_wait_tensorcnt(0);
#else
  asm volatile("s_wait_tensorcnt 0x0" ::: "memory");
#endif
#else
  asm volatile("s_wait_tensorcnt 0x0" ::: "memory");
#endif
}

// ---------------------------------------------------------------------------
// TDM: DMA one [rows=128] x [cols=32 halves] tile of a row-major f16 tensor
// (row stride = stride_elems) into LDS with +8-half row padding (-> LDT=40).
// D# per ISA 08_async §8.3/8.4:
//   g0: count=1 | lds_addr | global_addr[56:0] | type=2
//   g1: data_size=2B, pad_enable, pad_interval=3 (16 DW), pad_amount=3 (4 DW),
//       tensor_dim0/1, tile_dim0=32, tile_dim1=128, dim0_stride
// ---------------------------------------------------------------------------
#if HAVE_TDM
__device__ __forceinline__ void tdm_load_tile(const _Float16* gbase,
                                              unsigned lds_byte,
                                              unsigned tensor_d0,
                                              unsigned tensor_d1,
                                              unsigned stride_elems) {
  const unsigned long long ga = (unsigned long long)(uintptr_t)gbase;
  v4u g0 = {
      1u,                                                   // count=1 (valid)
      lds_byte,                                             // lds_addr
      (unsigned)(ga & 0xffffffffu),                         // global_addr lo
      (unsigned)((ga >> 32) & 0x01ffffffu) | (2u << 30)     // addr hi | type=2
  };
  const unsigned w0 = (1u << 16)      // data_size = 2 bytes
                    | (1u << 20)      // pad_enable
                    | (3u << 22)      // pad_interval: 16 DWORDs (64B of data)
                    | (3u << 25);     // pad_amount: 4 DWORDs (16B) -> LDT=40
  v8i g1 = {
      (int)w0,
      (int)((tensor_d0 & 0xffffu) << 16),                             // dim0 lo
      (int)(((tensor_d0 >> 16) & 0xffffu) | ((tensor_d1 & 0xffffu) << 16)),
      (int)(((tensor_d1 >> 16) & 0xffffu) | (32u << 16)),             // tile_dim0
      (int)128u,                                                      // tile_dim1
      (int)stride_elems,                                              // dim0 stride
      0, 0
  };
  v4i z4 = {0, 0, 0, 0};
#if __clang_major__ >= 23
  v8i z8 = {0, 0, 0, 0, 0, 0, 0, 0};
  __builtin_amdgcn_tensor_load_to_lds(g0, g1, z4, z4, z8, 0);
#else
  __builtin_amdgcn_tensor_load_to_lds(g0, g1, z4, z4, 0);
#endif
}
#endif

// ---------------------------------------------------------------------------
// f32 -> f16 convert
// ---------------------------------------------------------------------------
__global__ void __launch_bounds__(256) cvt_f32_f16(const float* __restrict__ s,
                                                   _Float16* __restrict__ d, int n) {
  int i = blockIdx.x * blockDim.x + threadIdx.x;
  if (i < n) d[i] = (_Float16)s[i];
}

// ---------------------------------------------------------------------------
// U1[n, p*512+f] = 0.25 * w1[p,f]*x[n,f] / max(||w1[p]*x[n]||, eps)   (f16)
// ---------------------------------------------------------------------------
__global__ void __launch_bounds__(256) u1_kernel(const float* __restrict__ x,
                                                 const float* __restrict__ w,
                                                 _Float16* __restrict__ U) {
  __shared__ float xs[DD_];
  __shared__ float red[256];
  const int n = blockIdx.x;
  const int t = threadIdx.x;
  xs[t]       = x[(size_t)n * DD_ + t];
  xs[t + 256] = x[(size_t)n * DD_ + t + 256];
  __syncthreads();
  for (int p = 0; p < PP_; ++p) {
    const float* wp = w + p * DD_;
    const float a0 = wp[t] * xs[t];
    const float a1 = wp[t + 256] * xs[t + 256];
    red[t] = a0 * a0 + a1 * a1;
    __syncthreads();
#pragma unroll
    for (int s = 128; s > 0; s >>= 1) {
      if (t < s) red[t] += red[t + s];
      __syncthreads();
    }
    const float sc = 0.25f / fmaxf(sqrtf(red[0]), EPS_);  // 0.25 = 1/sqrt(P)
    const size_t off = (size_t)n * K1_ + (size_t)p * DD_;
    U[off + t]       = (_Float16)(a0 * sc);
    U[off + t + 256] = (_Float16)(a1 * sc);
    __syncthreads();
  }
}

// ---------------------------------------------------------------------------
// U2 from concat([z1,z2]) : F = 128
// ---------------------------------------------------------------------------
__global__ void __launch_bounds__(128) u2_kernel(const float* __restrict__ z1,
                                                 const float* __restrict__ z2,
                                                 const float* __restrict__ w,
                                                 _Float16* __restrict__ U) {
  __shared__ float xs[2 * CC_];
  __shared__ float red[128];
  const int n = blockIdx.x;
  const int t = threadIdx.x;
  xs[t] = (t < CC_) ? z1[(size_t)n * CC_ + t] : z2[(size_t)n * CC_ + (t - CC_)];
  __syncthreads();
  for (int p = 0; p < PP_; ++p) {
    const float a = w[p * 2 * CC_ + t] * xs[t];
    red[t] = a * a;
    __syncthreads();
#pragma unroll
    for (int s = 64; s > 0; s >>= 1) {
      if (t < s) red[t] += red[t + s];
      __syncthreads();
    }
    const float sc = 0.25f / fmaxf(sqrtf(red[0]), EPS_);
    U[(size_t)n * K2_ + (size_t)p * 2 * CC_ + t] = (_Float16)(a * sc);
    __syncthreads();
  }
}

// ---------------------------------------------------------------------------
// Generic WMMA GEMM, double-buffered LDS pipeline.
//   C[M,N] = A[M,K] * B  (+bias, relu optional)
//   BT=true  : B row-major [N][K] (A*B^T Gram products) -> TDM tile DMA
//   BT=false : B row-major [K][N] -> per-lane async A + LDS scatter B
// 256 threads = 8 waves (wave32); 128x128 block tile; wave tile 64x32.
// ---------------------------------------------------------------------------
template <bool BT, bool RELU, bool BIAS>
__global__ void __launch_bounds__(256)
gemm_wmma(const _Float16* __restrict__ A, const _Float16* __restrict__ B,
          const float* __restrict__ bias, float* __restrict__ Cf,
          _Float16* __restrict__ Ch, int M, int N, int K, int lda, int ldb,
          int ldc) {
  __shared__ _Float16 As[2][BM * LDT];
  __shared__ _Float16 Bs[2][BN * LDT];

  const int tid  = threadIdx.x;
  const int lane = tid & 31;
  const int wave = tid >> 5;   // 0..7
  const int wm   = wave >> 2;  // 0..1 : 64-row band
  const int wn   = wave & 3;   // 0..3 : 32-col band
  const int l16  = lane & 15;
  const int lh   = lane >> 4;
  const int m0   = blockIdx.y * BM;
  const int n0   = blockIdx.x * BN;

  // per-thread staging coordinates: 2 threads per row, 16 halves (32B) each
  const int arow = tid >> 1;
  const int acol = (tid & 1) * 16;

  v8f acc[4][2];
#pragma unroll
  for (int mt = 0; mt < 4; ++mt)
#pragma unroll
    for (int nt = 0; nt < 2; ++nt)
      acc[mt][nt] = (v8f){0.f, 0.f, 0.f, 0.f, 0.f, 0.f, 0.f, 0.f};

  auto stage = [&](int buf, int kb) {
    if (BT) {
#if HAVE_TDM
      // one wave drives the Tensor Data Mover for both tiles
      if (wave == 0) {
        tdm_load_tile(A + (size_t)m0 * lda + kb,
                      (unsigned)(uintptr_t)&As[buf][0],
                      (unsigned)K, (unsigned)M, (unsigned)lda);
        tdm_load_tile(B + (size_t)n0 * ldb + kb,
                      (unsigned)(uintptr_t)&Bs[buf][0],
                      (unsigned)K, (unsigned)N, (unsigned)ldb);
      }
#else
      async_cp32(A + (size_t)(m0 + arow) * lda + kb + acol,
                 &As[buf][arow * LDT + acol]);
      int gr = n0 + arow;
      if (gr > N - 1) gr = N - 1;
      async_cp32(B + (size_t)gr * ldb + kb + acol,
                 &Bs[buf][arow * LDT + acol]);
#endif
    } else {
      // A tile: 128 rows x 32 halves, async DMA to LDS
      async_cp32(A + (size_t)(m0 + arow) * lda + kb + acol,
                 &As[buf][arow * LDT + acol]);
      // B row-major [K][N]: gather a 32x128 slab and scatter to [n][k] layout
      const int krow = tid >> 3;        // 0..31
      const int col  = (tid & 7) * 16;  // 0..112
      const _Float16* srow = B + (size_t)(kb + krow) * ldb;
      Pack32B u;
      if (n0 + col + 16 <= N) {
        u.q[0] = *(const uint4*)(srow + n0 + col);
        u.q[1] = *(const uint4*)(srow + n0 + col + 8);
      } else {
#pragma unroll
        for (int h = 0; h < 16; ++h) {
          int c = n0 + col + h;
          if (c > N - 1) c = N - 1;
          u.h[h] = srow[c];
        }
      }
#pragma unroll
      for (int h = 0; h < 16; ++h) Bs[buf][(col + h) * LDT + krow] = u.h[h];
    }
  };

  const int nk = K / BK;
  int buf = 0;
  stage(0, 0);

  for (int ki = 0; ki < nk; ++ki) {
    if (BT) {
#if HAVE_TDM
      wait_tensor0();  // wave 0: its TDM tiles have landed; others: no-op
#else
      wait_async0();
#endif
    } else {
      wait_async0();   // own async writes into `buf` have landed in LDS
    }
    __syncthreads();   // ... and everyone's writes are published
    if (ki + 1 < nk) stage(buf ^ 1, (ki + 1) * BK);  // overlap next tile

    // ---- build fragments (CDNA5 16-bit A 16x32 / B 32x16 layouts) ----
    AFragU af[4];
#pragma unroll
    for (int mt = 0; mt < 4; ++mt) {
      const _Float16* base = &As[buf][(wm * 64 + mt * 16 + l16) * LDT];
#pragma unroll
      for (int v = 0; v < 8; ++v) {
        const int ko = 2 * v + 16 * lh;
        af[mt].u[v] = *(const unsigned int*)(base + ko);
      }
    }
    BFragU bf[2];
#pragma unroll
    for (int nt = 0; nt < 2; ++nt) {
      const _Float16* base = &Bs[buf][(wn * 32 + nt * 16 + l16) * LDT + lh * 16];
      bf[nt].q[0] = *(const uint4*)(base);
      bf[nt].q[1] = *(const uint4*)(base + 8);
    }
    // ---- 8 WMMAs per k-step per wave ----
#pragma unroll
    for (int mt = 0; mt < 4; ++mt)
#pragma unroll
      for (int nt = 0; nt < 2; ++nt)
        acc[mt][nt] = __builtin_amdgcn_wmma_f32_16x16x32_f16(
            false, af[mt].v, false, bf[nt].v, (short)0, acc[mt][nt], false, false);
    buf ^= 1;
  }

  // ---- epilogue: C 16x16 f32 layout -> row = base + r + 8*lh, col = l16 ----
#pragma unroll
  for (int nt = 0; nt < 2; ++nt) {
    const int col = n0 + wn * 32 + nt * 16 + l16;
    float bv = 0.f;
    if (BIAS) bv = (col < N) ? bias[col] : 0.f;
#pragma unroll
    for (int mt = 0; mt < 4; ++mt) {
      const int rowb = m0 + wm * 64 + mt * 16 + lh * 8;
#pragma unroll
      for (int r = 0; r < 8; ++r) {
        float v = acc[mt][nt][r];
        if (BIAS) v += bv;
        if (RELU) v = v > 0.f ? v : 0.f;
        if (col < N) {
          const size_t idx = (size_t)(rowb + r) * ldc + col;
          if (Cf) Cf[idx] = v;
          if (Ch) Ch[idx] = (_Float16)v;
        }
      }
    }
  }
}

// ---------------------------------------------------------------------------
// adj1 row-normalize in place (att already relu'd), also emit f16 copy
// ---------------------------------------------------------------------------
__global__ void __launch_bounds__(256) rownorm_adj1(float* __restrict__ att,
                                                    _Float16* __restrict__ adjh) {
  __shared__ float red[256];
  const int n = blockIdx.x;
  const int t = threadIdx.x;
  float s = 0.f;
  for (int c = t; c < NN_; c += 256) s += att[(size_t)n * NN_ + c];
  red[t] = s;
  __syncthreads();
#pragma unroll
  for (int k = 128; k > 0; k >>= 1) {
    if (t < k) red[t] += red[t + k];
    __syncthreads();
  }
  const float inv = 1.f / fmaxf(red[0], EPS_);
  for (int c = t; c < NN_; c += 256) {
    const size_t i = (size_t)n * NN_ + c;
    const float v = att[i] * inv;
    att[i]  = v;
    adjh[i] = (_Float16)v;
  }
}

// ---------------------------------------------------------------------------
// adjOut (holds relu'd att2) -> 0.5*adj1 + 0.5*att2/rowsum, in place
// ---------------------------------------------------------------------------
__global__ void __launch_bounds__(256) final_adj(float* __restrict__ adjOut,
                                                 const float* __restrict__ adj1) {
  __shared__ float red[256];
  const int n = blockIdx.x;
  const int t = threadIdx.x;
  float s = 0.f;
  for (int c = t; c < NN_; c += 256) s += adjOut[(size_t)n * NN_ + c];
  red[t] = s;
  __syncthreads();
#pragma unroll
  for (int k = 128; k > 0; k >>= 1) {
    if (t < k) red[t] += red[t + k];
    __syncthreads();
  }
  const float hinv = 0.5f / fmaxf(red[0], EPS_);
  for (int c = t; c < NN_; c += 256) {
    const size_t i = (size_t)n * NN_ + c;
    adjOut[i] = 0.5f * adj1[i] + hinv * adjOut[i];
  }
}

// ---------------------------------------------------------------------------
// output = 0.5*(z1+z2)
// ---------------------------------------------------------------------------
__global__ void __launch_bounds__(256) mix_out(const float* __restrict__ z1,
                                               const float* __restrict__ z2,
                                               float* __restrict__ o, int n) {
  int i = blockIdx.x * blockDim.x + threadIdx.x;
  if (i < n) o[i] = 0.5f * (z1[i] + z2[i]);
}

// ---------------------------------------------------------------------------
// launcher
// ---------------------------------------------------------------------------
extern "C" void kernel_launch(void* const* d_in, const int* in_sizes, int n_in,
                              void* d_out, int out_size, void* d_ws,
                              size_t ws_size, hipStream_t stream) {
  const float* feats = (const float*)d_in[0];
  const float* w_gl1 = (const float*)d_in[1];
  const float* w_gl2 = (const float*)d_in[2];
  const float* Wg1   = (const float*)d_in[3];
  const float* bg1   = (const float*)d_in[4];
  const float* Wg2   = (const float*)d_in[5];
  const float* bg2   = (const float*)d_in[6];
  const float* Wm1   = (const float*)d_in[7];
  const float* bm1   = (const float*)d_in[8];
  const float* Wm2   = (const float*)d_in[9];
  const float* bm2   = (const float*)d_in[10];

  float* out    = (float*)d_out;            // [4096,64]
  float* adjOut = out + (size_t)NN_ * CC_;  // [4096,4096]

  // ---- workspace carve (256B aligned) ----
  size_t off = 0;
  char* base = (char*)d_ws;
  auto carve = [&](size_t bytes) -> void* {
    void* p = base + off;
    off += (bytes + 255) & ~(size_t)255;
    return p;
  };
  _Float16* featsh = (_Float16*)carve((size_t)NN_ * DD_ * 2);
  _Float16* Wg1h   = (_Float16*)carve((size_t)DD_ * NHID_ * 2);
  _Float16* Wg2h   = (_Float16*)carve((size_t)NHID_ * CC_ * 2);
  _Float16* Wm1h   = (_Float16*)carve((size_t)DD_ * NHID_ * 2);
  _Float16* Wm2h   = (_Float16*)carve((size_t)NHID_ * CC_ * 2);
  _Float16* U1h    = (_Float16*)carve((size_t)NN_ * K1_ * 2);
  float*    adj1f  = (float*)   carve((size_t)NN_ * NN_ * 4);  // att1 then adj1
  _Float16* adj1h  = (_Float16*)carve((size_t)NN_ * NN_ * 2);
  _Float16* Xg1h   = (_Float16*)carve((size_t)NN_ * NHID_ * 2);
  _Float16* hh     = (_Float16*)carve((size_t)NN_ * NHID_ * 2);
  _Float16* t2h    = (_Float16*)carve((size_t)NN_ * CC_ * 2);
  float*    z1     = (float*)   carve((size_t)NN_ * CC_ * 4);
  float*    z2     = (float*)   carve((size_t)NN_ * CC_ * 4);
  _Float16* m1h    = (_Float16*)carve((size_t)NN_ * NHID_ * 2);
  _Float16* U2h    = (_Float16*)carve((size_t)NN_ * K2_ * 2);
  (void)ws_size; (void)in_sizes; (void)n_in; (void)out_size;

  const dim3 blk(256);
  auto g1 = [](int n) { return dim3((n + 255) / 256); };

  // 1) f16 conversions
  cvt_f32_f16<<<g1(NN_ * DD_), blk, 0, stream>>>(feats, featsh, NN_ * DD_);
  cvt_f32_f16<<<g1(DD_ * NHID_), blk, 0, stream>>>(Wg1, Wg1h, DD_ * NHID_);
  cvt_f32_f16<<<g1(NHID_ * CC_), blk, 0, stream>>>(Wg2, Wg2h, NHID_ * CC_);
  cvt_f32_f16<<<g1(DD_ * NHID_), blk, 0, stream>>>(Wm1, Wm1h, DD_ * NHID_);
  cvt_f32_f16<<<g1(NHID_ * CC_), blk, 0, stream>>>(Wm2, Wm2h, NHID_ * CC_);

  // 2) U1 (normalized per-head scaled feats, 1/sqrt(P) folded in)
  u1_kernel<<<dim3(NN_), blk, 0, stream>>>(feats, w_gl1, U1h);

  // 3) att1 = relu(U1 U1^T)   [dominant GEMM: 4096x4096x8192]
  gemm_wmma<true, true, false><<<dim3(NN_ / BN, NN_ / BM), blk, 0, stream>>>(
      U1h, U1h, nullptr, adj1f, nullptr, NN_, NN_, K1_, K1_, K1_, NN_);

  // 4) adj1 = rownorm(att1)  (in place) + f16 copy
  rownorm_adj1<<<dim3(NN_), blk, 0, stream>>>(adj1f, adj1h);

  // 5) Xg1 = feats @ Wg1            (f16 out)
  gemm_wmma<false, false, false><<<dim3(NHID_ / BN, NN_ / BM), blk, 0, stream>>>(
      featsh, Wg1h, nullptr, nullptr, Xg1h, NN_, NHID_, DD_, DD_, NHID_, NHID_);

  // 6) h = relu(adj1 @ Xg1 + bg1)   (f16 out)
  gemm_wmma<false, true, true><<<dim3(NHID_ / BN, NN_ / BM), blk, 0, stream>>>(
      adj1h, Xg1h, bg1, nullptr, hh, NN_, NHID_, NN_, NN_, NHID_, NHID_);

  // 7) t2 = h @ Wg2                 (f16 out; N=64 handled by guards)
  gemm_wmma<false, false, false><<<dim3(1, NN_ / BM), blk, 0, stream>>>(
      hh, Wg2h, nullptr, nullptr, t2h, NN_, CC_, NHID_, NHID_, CC_, CC_);

  // 8) z1 = adj1 @ t2 + bg2         (f32 out)
  gemm_wmma<false, false, true><<<dim3(1, NN_ / BM), blk, 0, stream>>>(
      adj1h, t2h, bg2, z1, nullptr, NN_, CC_, NN_, NN_, CC_, CC_);

  // 9) m1 = relu(feats @ Wm1 + bm1) (f16 out)
  gemm_wmma<false, true, true><<<dim3(NHID_ / BN, NN_ / BM), blk, 0, stream>>>(
      featsh, Wm1h, bm1, nullptr, m1h, NN_, NHID_, DD_, DD_, NHID_, NHID_);

  // 10) z2 = m1 @ Wm2 + bm2         (f32 out)
  gemm_wmma<false, false, true><<<dim3(1, NN_ / BM), blk, 0, stream>>>(
      m1h, Wm2h, bm2, z2, nullptr, NN_, CC_, NHID_, NHID_, CC_, CC_);

  // 11) U2 from concat(z1, z2)
  u2_kernel<<<dim3(NN_), dim3(128), 0, stream>>>(z1, z2, w_gl2, U2h);

  // 12) att2 = relu(U2 U2^T) directly into d_out adj region
  gemm_wmma<true, true, false><<<dim3(NN_ / BN, NN_ / BM), blk, 0, stream>>>(
      U2h, U2h, nullptr, adjOut, nullptr, NN_, NN_, K2_, K2_, K2_, NN_);

  // 13) adj = 0.5*adj1 + 0.5*rownorm(att2)  (in place in d_out)
  final_adj<<<dim3(NN_), blk, 0, stream>>>(adjOut, adj1f);

  // 14) output = 0.5*(z1+z2)
  mix_out<<<g1(NN_ * CC_), blk, 0, stream>>>(z1, z2, out, NN_ * CC_);
}
